// EGNN_dynamics_QM9_34411277975641
// MI455X (gfx1250) — compile-verified, matrix operand order checked by reading, not run on hardware
//
#include <hip/hip_runtime.h>
#include <hip/hip_bf16.h>
#include <stdint.h>

// ---------------------------------------------------------------------------
// EGNN forward for MI455X (gfx1250, wave32, WMMA 16x16x32 f16).
// Heavy edge/coord MLPs run as fused two-layer WMMA GEMMs, one workgroup per
// (batch, node-i) owning all 64 j-edges, with LDS staging + in-LDS reduction.
// ---------------------------------------------------------------------------

typedef _Float16 h8v  __attribute__((ext_vector_type(8)));
typedef _Float16 h16v __attribute__((ext_vector_type(16)));
typedef float    f8v  __attribute__((ext_vector_type(8)));

#define LDS_STRIDE 264   // 256 + 8 halves padding (keeps 16B alignment)

__device__ __forceinline__ h16v cat8(h8v lo, h8v hi) {
  return __builtin_shufflevector(lo, hi, 0,1,2,3,4,5,6,7,8,9,10,11,12,13,14,15);
}

// A-fragment (16x32 f16): base points at row_data + kc*32. Per ISA layout,
// lanes 0-15 carry K {0..7, 16..23}; lanes 16-31 carry K {8..15, 24..31}.
__device__ __forceinline__ h16v frag_row(const _Float16* base, int lane) {
  int off = (lane >> 4) << 3;                      // 0 or 8
  h8v lo = *(const h8v*)(base + off);              // K off..off+7
  h8v hi = *(const h8v*)(base + off + 16);         // K off+16..off+23
  return cat8(lo, hi);
}

// B-fragment from repacked weights: [kc][nt][lane][16] contiguous halves.
__device__ __forceinline__ h16v frag_b(const _Float16* w, int kc, int nt, int lane) {
  const _Float16* p = w + (size_t)((((kc << 4) + nt) << 5) + lane) * 16;
  h8v lo = *(const h8v*)(p);
  h8v hi = *(const h8v*)(p + 8);
  return cat8(lo, hi);
}

__device__ __forceinline__ f8v wmma16(h16v a, h16v b, f8v c) {
  return __builtin_amdgcn_wmma_f32_16x16x32_f16(false, a, false, b, (short)0, c, false, false);
}

__device__ __forceinline__ float silu(float x) {
  return x * (1.0f / (1.0f + __expf(-x)));
}

__device__ __forceinline__ f8v zero8() {
  f8v z = {0.f,0.f,0.f,0.f,0.f,0.f,0.f,0.f};
  return z;
}

// ---------------------------------------------------------------------------
// Weight repack: f32 [K x 256] row-major -> f16 fragment-ready layout.
// ---------------------------------------------------------------------------
__global__ void repack_kernel(const float* __restrict__ w, _Float16* __restrict__ out, int KC) {
  int idx = blockIdx.x * 256 + threadIdx.x;
  int total = KC * 16 * 32 * 16;
  if (idx >= total) return;
  int hh   = idx & 15;
  int lane = (idx >> 4) & 31;
  int nt   = (idx >> 9) & 15;
  int kc   = idx >> 13;
  int off  = (lane >> 4) << 3;
  int nn   = lane & 15;
  int kk   = off + (hh < 8 ? hh : hh + 8);
  out[idx] = (_Float16)w[(size_t)(kc * 32 + kk) * 256 + nt * 16 + nn];
}

__global__ void zero_kernel(float* p, int n) {
  int i = blockIdx.x * 256 + threadIdx.x;
  if (i < n) p[i] = 0.f;
}

// ---------------------------------------------------------------------------
// Embed: h = [xh[...,3:]*nm, t] @ We + be ; also stash masked x0 and x.
// ---------------------------------------------------------------------------
__global__ __launch_bounds__(256) void embed_kernel(
    const float* __restrict__ t, const float* __restrict__ xh,
    const float* __restrict__ nm, const float* __restrict__ we,
    const float* __restrict__ be, float* __restrict__ hf,
    _Float16* __restrict__ h16, float* __restrict__ x0, float* __restrict__ xA) {
  int row = blockIdx.x;            // 0..2047  (b*64 + n)
  int b   = row >> 6;
  int c   = threadIdx.x;           // 0..255
  float nmv = nm[row];
  float in[7];
#pragma unroll
  for (int k = 0; k < 6; ++k) in[k] = xh[row * 9 + 3 + k] * nmv;
  in[6] = t[b];
  float s = be[c];
#pragma unroll
  for (int k = 0; k < 7; ++k) s += in[k] * we[k * 256 + c];
  size_t gi = (size_t)row * 256 + c;
  hf[gi]  = s;
  h16[gi] = (_Float16)s;
  if (c < 3) {
    float v = xh[row * 9 + c] * nmv;
    x0[row * 3 + c] = v;
    xA[row * 3 + c] = v;
  }
}

// ---------------------------------------------------------------------------
// Fused edge MLP + aggregation: one workgroup per (b,i), 8 waves.
// edge1(K=512 + rank-2 tail) -> SiLU -> edge2(K=256) -> SiLU -> *em -> sum_j/100
// ---------------------------------------------------------------------------
__global__ __launch_bounds__(256) void edge_gcl_kernel(
    const _Float16* __restrict__ h16, const float* __restrict__ xcur,
    const float* __restrict__ x0, const float* __restrict__ em,
    const _Float16* __restrict__ w1r, const float* __restrict__ w1tail,
    const float* __restrict__ b1,
    const _Float16* __restrict__ w2r, const float* __restrict__ b2,
    _Float16* __restrict__ agg16) {
  __shared__ __align__(16) _Float16 e1[64 * LDS_STRIDE];
  __shared__ float dl[64], d0[64], emv[64], aggl[256];
  int tid = threadIdx.x;
  int bi = blockIdx.x, b = bi >> 6, i = bi & 63;

  aggl[tid] = 0.f;
  if (tid < 64) {
    int j = tid;
    float s = 0.f, s0 = 0.f;
#pragma unroll
    for (int d = 0; d < 3; ++d) {
      float dd = xcur[(b * 64 + i) * 3 + d] - xcur[(b * 64 + j) * 3 + d];
      s += dd * dd;
      float e0 = x0[(b * 64 + i) * 3 + d] - x0[(b * 64 + j) * 3 + d];
      s0 += e0 * e0;
    }
    dl[j] = s; d0[j] = s0;
    emv[j] = em[(size_t)(b * 64 + i) * 64 + j];
  }
  __syncthreads();

  int wave = tid >> 5, lane = tid & 31;
  int n15 = lane & 15, mh = (lane >> 4) << 3;
  const _Float16* hb = h16 + (size_t)b * 64 * 256;

  f8v acc[4][2];
#pragma unroll
  for (int mt = 0; mt < 4; ++mt) { acc[mt][0] = zero8(); acc[mt][1] = zero8(); }

  for (int kc = 0; kc < 16; ++kc) {
    h16v B0 = frag_b(w1r, kc, wave * 2 + 0, lane);
    h16v B1 = frag_b(w1r, kc, wave * 2 + 1, lane);
    if (kc < 8) {                         // h_i columns: broadcast A
      h16v A = frag_row(hb + (size_t)i * 256 + kc * 32, lane);
#pragma unroll
      for (int mt = 0; mt < 4; ++mt) {
        acc[mt][0] = wmma16(A, B0, acc[mt][0]);
        acc[mt][1] = wmma16(A, B1, acc[mt][1]);
      }
    } else {                              // h_j columns
#pragma unroll
      for (int mt = 0; mt < 4; ++mt) {
        int j = mt * 16 + n15;
        h16v A = frag_row(hb + (size_t)j * 256 + (kc - 8) * 32, lane);
        acc[mt][0] = wmma16(A, B0, acc[mt][0]);
        acc[mt][1] = wmma16(A, B1, acc[mt][1]);
      }
    }
  }

  // epilogue 1: rank-2 tail (d_l, dist0) + bias + SiLU -> LDS f16
#pragma unroll
  for (int ct = 0; ct < 2; ++ct) {
    int n = (wave * 2 + ct) * 16 + n15;
    float wr0 = w1tail[n], wr1 = w1tail[256 + n], bb = b1[n];
#pragma unroll
    for (int mt = 0; mt < 4; ++mt) {
#pragma unroll
      for (int v = 0; v < 8; ++v) {
        int j = mt * 16 + v + mh;
        float a = acc[mt][ct][v] + dl[j] * wr0 + d0[j] * wr1 + bb;
        e1[j * LDS_STRIDE + n] = (_Float16)silu(a);
      }
    }
  }
  __syncthreads();

  // edge2: K=256 from LDS
  f8v acc2[4][2];
#pragma unroll
  for (int mt = 0; mt < 4; ++mt) { acc2[mt][0] = zero8(); acc2[mt][1] = zero8(); }
  for (int kc = 0; kc < 8; ++kc) {
    h16v B0 = frag_b(w2r, kc, wave * 2 + 0, lane);
    h16v B1 = frag_b(w2r, kc, wave * 2 + 1, lane);
#pragma unroll
    for (int mt = 0; mt < 4; ++mt) {
      h16v A = frag_row(&e1[(mt * 16 + n15) * LDS_STRIDE + kc * 32], lane);
      acc2[mt][0] = wmma16(A, B0, acc2[mt][0]);
      acc2[mt][1] = wmma16(A, B1, acc2[mt][1]);
    }
  }

  // epilogue 2: bias + SiLU + edge mask, reduce over j into LDS
#pragma unroll
  for (int ct = 0; ct < 2; ++ct) {
    int n = (wave * 2 + ct) * 16 + n15;
    float bb = b2[n];
    float part = 0.f;
#pragma unroll
    for (int mt = 0; mt < 4; ++mt) {
#pragma unroll
      for (int v = 0; v < 8; ++v) {
        int j = mt * 16 + v + mh;
        part += silu(acc2[mt][ct][v] + bb) * emv[j];
      }
    }
    atomicAdd(&aggl[n], part);
  }
  __syncthreads();
  agg16[((size_t)b * 64 + i) * 256 + tid] = (_Float16)(aggl[tid] * 0.01f);
}

// ---------------------------------------------------------------------------
// Node MLP: upd = node2(silu(node1([h, agg]))); h = (h + upd) * nm.
// One workgroup per batch b (M = 64 rows).
// ---------------------------------------------------------------------------
__global__ __launch_bounds__(256) void node_gcl_kernel(
    const _Float16* h16in, const _Float16* __restrict__ agg16,
    const float* __restrict__ nm,
    const _Float16* __restrict__ w1r, const float* __restrict__ b1,
    const _Float16* __restrict__ w2r, const float* __restrict__ b2,
    float* hf, _Float16* h16out) {
  __shared__ __align__(16) _Float16 u1[64 * LDS_STRIDE];
  int tid = threadIdx.x, b = blockIdx.x;
  int wave = tid >> 5, lane = tid & 31;
  int n15 = lane & 15, mh = (lane >> 4) << 3;
  const _Float16* hb = h16in + (size_t)b * 64 * 256;
  const _Float16* ab = agg16 + (size_t)b * 64 * 256;

  f8v acc[4][2];
#pragma unroll
  for (int mt = 0; mt < 4; ++mt) { acc[mt][0] = zero8(); acc[mt][1] = zero8(); }
  for (int kc = 0; kc < 16; ++kc) {
    h16v B0 = frag_b(w1r, kc, wave * 2 + 0, lane);
    h16v B1 = frag_b(w1r, kc, wave * 2 + 1, lane);
#pragma unroll
    for (int mt = 0; mt < 4; ++mt) {
      int row = mt * 16 + n15;
      const _Float16* src = (kc < 8) ? (hb + (size_t)row * 256 + kc * 32)
                                     : (ab + (size_t)row * 256 + (kc - 8) * 32);
      h16v A = frag_row(src, lane);
      acc[mt][0] = wmma16(A, B0, acc[mt][0]);
      acc[mt][1] = wmma16(A, B1, acc[mt][1]);
    }
  }
#pragma unroll
  for (int ct = 0; ct < 2; ++ct) {
    int n = (wave * 2 + ct) * 16 + n15;
    float bb = b1[n];
#pragma unroll
    for (int mt = 0; mt < 4; ++mt) {
#pragma unroll
      for (int v = 0; v < 8; ++v) {
        int row = mt * 16 + v + mh;
        u1[row * LDS_STRIDE + n] = (_Float16)silu(acc[mt][ct][v] + bb);
      }
    }
  }
  __syncthreads();

  f8v acc2[4][2];
#pragma unroll
  for (int mt = 0; mt < 4; ++mt) { acc2[mt][0] = zero8(); acc2[mt][1] = zero8(); }
  for (int kc = 0; kc < 8; ++kc) {
    h16v B0 = frag_b(w2r, kc, wave * 2 + 0, lane);
    h16v B1 = frag_b(w2r, kc, wave * 2 + 1, lane);
#pragma unroll
    for (int mt = 0; mt < 4; ++mt) {
      h16v A = frag_row(&u1[(mt * 16 + n15) * LDS_STRIDE + kc * 32], lane);
      acc2[mt][0] = wmma16(A, B0, acc2[mt][0]);
      acc2[mt][1] = wmma16(A, B1, acc2[mt][1]);
    }
  }
#pragma unroll
  for (int ct = 0; ct < 2; ++ct) {
    int n = (wave * 2 + ct) * 16 + n15;
    float bb = b2[n];
#pragma unroll
    for (int mt = 0; mt < 4; ++mt) {
#pragma unroll
      for (int v = 0; v < 8; ++v) {
        int row = mt * 16 + v + mh;
        size_t gi = ((size_t)b * 64 + row) * 256 + n;
        float nmv = nm[b * 64 + row];
        float hn = (hf[gi] + acc2[mt][ct][v] + bb) * nmv;
        hf[gi] = hn;
        h16out[gi] = (_Float16)hn;
      }
    }
  }
}

// ---------------------------------------------------------------------------
// Coord MLP: phi = coord3(silu(coord2(silu(coord1(eij)))));
// x_next = (x + sum_j coord_diff*phi*em / 100) * nm. One WG per (b,i).
// ---------------------------------------------------------------------------
__global__ __launch_bounds__(256) void coord_kernel(
    const _Float16* __restrict__ h16, const float* __restrict__ xcur,
    const float* __restrict__ x0, const float* __restrict__ em,
    const float* __restrict__ nm,
    const _Float16* __restrict__ w1r, const float* __restrict__ w1tail,
    const float* __restrict__ b1,
    const _Float16* __restrict__ w2r, const float* __restrict__ b2,
    const float* __restrict__ w3, float* __restrict__ xnext) {
  __shared__ __align__(16) _Float16 e1[64 * LDS_STRIDE];   // reused for layer-2 out
  __shared__ float dl[64], d0[64], emv[64], phi[64];
  int tid = threadIdx.x;
  int bi = blockIdx.x, b = bi >> 6, i = bi & 63;

  if (tid < 64) {
    int j = tid;
    float s = 0.f, s0 = 0.f;
#pragma unroll
    for (int d = 0; d < 3; ++d) {
      float dd = xcur[(b * 64 + i) * 3 + d] - xcur[(b * 64 + j) * 3 + d];
      s += dd * dd;
      float e0 = x0[(b * 64 + i) * 3 + d] - x0[(b * 64 + j) * 3 + d];
      s0 += e0 * e0;
    }
    dl[j] = s; d0[j] = s0;
    emv[j] = em[(size_t)(b * 64 + i) * 64 + j];
    phi[j] = 0.f;
  }
  __syncthreads();

  int wave = tid >> 5, lane = tid & 31;
  int n15 = lane & 15, mh = (lane >> 4) << 3;
  const _Float16* hb = h16 + (size_t)b * 64 * 256;

  f8v acc[4][2];
#pragma unroll
  for (int mt = 0; mt < 4; ++mt) { acc[mt][0] = zero8(); acc[mt][1] = zero8(); }
  for (int kc = 0; kc < 16; ++kc) {
    h16v B0 = frag_b(w1r, kc, wave * 2 + 0, lane);
    h16v B1 = frag_b(w1r, kc, wave * 2 + 1, lane);
    if (kc < 8) {
      h16v A = frag_row(hb + (size_t)i * 256 + kc * 32, lane);
#pragma unroll
      for (int mt = 0; mt < 4; ++mt) {
        acc[mt][0] = wmma16(A, B0, acc[mt][0]);
        acc[mt][1] = wmma16(A, B1, acc[mt][1]);
      }
    } else {
#pragma unroll
      for (int mt = 0; mt < 4; ++mt) {
        int j = mt * 16 + n15;
        h16v A = frag_row(hb + (size_t)j * 256 + (kc - 8) * 32, lane);
        acc[mt][0] = wmma16(A, B0, acc[mt][0]);
        acc[mt][1] = wmma16(A, B1, acc[mt][1]);
      }
    }
  }
#pragma unroll
  for (int ct = 0; ct < 2; ++ct) {
    int n = (wave * 2 + ct) * 16 + n15;
    float wr0 = w1tail[n], wr1 = w1tail[256 + n], bb = b1[n];
#pragma unroll
    for (int mt = 0; mt < 4; ++mt) {
#pragma unroll
      for (int v = 0; v < 8; ++v) {
        int j = mt * 16 + v + mh;
        float a = acc[mt][ct][v] + dl[j] * wr0 + d0[j] * wr1 + bb;
        e1[j * LDS_STRIDE + n] = (_Float16)silu(a);
      }
    }
  }
  __syncthreads();

  f8v acc2[4][2];
#pragma unroll
  for (int mt = 0; mt < 4; ++mt) { acc2[mt][0] = zero8(); acc2[mt][1] = zero8(); }
  for (int kc = 0; kc < 8; ++kc) {
    h16v B0 = frag_b(w2r, kc, wave * 2 + 0, lane);
    h16v B1 = frag_b(w2r, kc, wave * 2 + 1, lane);
#pragma unroll
    for (int mt = 0; mt < 4; ++mt) {
      h16v A = frag_row(&e1[(mt * 16 + n15) * LDS_STRIDE + kc * 32], lane);
      acc2[mt][0] = wmma16(A, B0, acc2[mt][0]);
      acc2[mt][1] = wmma16(A, B1, acc2[mt][1]);
    }
  }
  __syncthreads();   // all reads of e1 done; safe to overwrite with layer-2 output
#pragma unroll
  for (int ct = 0; ct < 2; ++ct) {
    int n = (wave * 2 + ct) * 16 + n15;
    float bb = b2[n];
#pragma unroll
    for (int mt = 0; mt < 4; ++mt) {
#pragma unroll
      for (int v = 0; v < 8; ++v) {
        int j = mt * 16 + v + mh;
        e1[j * LDS_STRIDE + n] = (_Float16)silu(acc2[mt][ct][v] + bb);
      }
    }
  }
  __syncthreads();

  // coord3 matvec: phi[j] = sum_n e2[j][n] * w3[n]
  {
    int j = tid >> 2, q = tid & 3;
    float part = 0.f;
#pragma unroll 4
    for (int n = q * 64; n < q * 64 + 64; ++n)
      part += (float)e1[j * LDS_STRIDE + n] * w3[n];
    atomicAdd(&phi[j], part);
  }
  __syncthreads();

  if (tid < 3) {
    int d = tid;
    float xi = xcur[(b * 64 + i) * 3 + d];
    float s = 0.f;
    for (int j = 0; j < 64; ++j) {
      float cd = (xi - xcur[(b * 64 + j) * 3 + d]) * __frsqrt_rn(dl[j] + 1e-8f);
      s += cd * phi[j] * emv[j];
    }
    xnext[(b * 64 + i) * 3 + d] = (xi + s * 0.01f) * nm[b * 64 + i];
  }
}

// ---------------------------------------------------------------------------
// Final stage 1: h_out = embed_out(h)*nm ; vel_raw = (x - x0)*nm ; reductions.
// ---------------------------------------------------------------------------
__global__ __launch_bounds__(32) void final1_kernel(
    const float* __restrict__ hf, const float* __restrict__ wo,
    const float* __restrict__ bo, const float* __restrict__ nm,
    const float* __restrict__ xfin, const float* __restrict__ x0,
    float* __restrict__ hobuf, float* __restrict__ velraw,
    float* velsum, float* ncount) {
  int row = blockIdx.x, b = row >> 6, tid = threadIdx.x;
  float nmv = nm[row];
  if (tid < 7) {
    float s = bo[tid];
    for (int k = 0; k < 256; ++k) s += hf[(size_t)row * 256 + k] * wo[k * 7 + tid];
    hobuf[row * 7 + tid] = s * nmv;
  } else if (tid < 10) {
    int d = tid - 7;
    float v = (xfin[row * 3 + d] - x0[row * 3 + d]) * nmv;
    velraw[row * 3 + d] = v;
    atomicAdd(&velsum[b * 4 + d], v);
  } else if (tid == 10) {
    atomicAdd(&ncount[b], nmv);
  }
}

__global__ __launch_bounds__(32) void final2_kernel(
    const float* __restrict__ hobuf, const float* __restrict__ velraw,
    const float* __restrict__ velsum, const float* __restrict__ ncount,
    const float* __restrict__ nm, float* __restrict__ out) {
  int row = blockIdx.x, b = row >> 6, tid = threadIdx.x;
  if (tid < 3) {
    out[row * 9 + tid] = velraw[row * 3 + tid] - velsum[b * 4 + tid] / ncount[b] * nm[row];
  } else if (tid < 9) {
    out[row * 9 + tid] = hobuf[row * 7 + (tid - 3)];   // h[..., :-1]
  }
}

// ---------------------------------------------------------------------------
// Host: parse flattened inputs, carve workspace, run the pipeline on `stream`.
// Param leaf order assumes pytree flatten (sorted dict keys, lists in order):
// blocks[4]{coord1.b,w; coord2.b,w; coord3.w; gcls[2]{edge1.b,w; edge2.b,w;
// node1.b,w; node2.b,w}}, embed.b,w, embed_out.b,w.
// ---------------------------------------------------------------------------
extern "C" void kernel_launch(void* const* d_in, const int* in_sizes, int n_in,
                              void* d_out, int out_size, void* d_ws, size_t ws_size,
                              hipStream_t stream) {
  (void)in_sizes; (void)n_in; (void)out_size; (void)ws_size;
  const float* t  = (const float*)d_in[0];
  const float* xh = (const float*)d_in[1];
  const float* nm = (const float*)d_in[2];
  const float* em = (const float*)d_in[3];

  const float *c1b[4], *c1w[4], *c2b[4], *c2w[4], *c3w[4];
  const float *e1b[4][2], *e1w[4][2], *e2b[4][2], *e2w[4][2];
  const float *n1b[4][2], *n1w[4][2], *n2b[4][2], *n2w[4][2];
  int c = 4;
  for (int blk = 0; blk < 4; ++blk) {
    c1b[blk] = (const float*)d_in[c++]; c1w[blk] = (const float*)d_in[c++];
    c2b[blk] = (const float*)d_in[c++]; c2w[blk] = (const float*)d_in[c++];
    c3w[blk] = (const float*)d_in[c++];
    for (int g = 0; g < 2; ++g) {
      e1b[blk][g] = (const float*)d_in[c++]; e1w[blk][g] = (const float*)d_in[c++];
      e2b[blk][g] = (const float*)d_in[c++]; e2w[blk][g] = (const float*)d_in[c++];
      n1b[blk][g] = (const float*)d_in[c++]; n1w[blk][g] = (const float*)d_in[c++];
      n2b[blk][g] = (const float*)d_in[c++]; n2w[blk][g] = (const float*)d_in[c++];
    }
  }
  const float* emb_b  = (const float*)d_in[c++];
  const float* emb_w  = (const float*)d_in[c++];
  const float* embo_b = (const float*)d_in[c++];
  const float* embo_w = (const float*)d_in[c++];

  // Workspace carve (aligned to 256B).
  char* p = (char*)d_ws;
  auto carve = [&](size_t bytes) -> void* {
    void* r = (void*)p;
    p += (bytes + 255) & ~(size_t)255;
    return r;
  };
  const size_t ROWS = 32 * 64;                       // 2048
  float*    hf    = (float*)   carve(ROWS * 256 * 4);
  _Float16* h16   = (_Float16*)carve(ROWS * 256 * 2);
  _Float16* agg16 = (_Float16*)carve(ROWS * 256 * 2);
  float*    x0    = (float*)   carve(ROWS * 3 * 4);
  float*    xA    = (float*)   carve(ROWS * 3 * 4);
  float*    xB    = (float*)   carve(ROWS * 3 * 4);
  float*    velraw= (float*)   carve(ROWS * 3 * 4);
  float*    velsum= (float*)   carve(160 * 4);       // [128] velsum + [32] ncount
  float*    ncnt  = velsum + 128;
  float*    hobuf = (float*)   carve(ROWS * 7 * 4);
  _Float16 *re1[4][2], *re2[4][2], *rn1[4][2], *rn2[4][2], *rc1[4], *rc2[4];
  for (int blk = 0; blk < 4; ++blk) {
    rc1[blk] = (_Float16*)carve(512 * 256 * 2);
    rc2[blk] = (_Float16*)carve(256 * 256 * 2);
    for (int g = 0; g < 2; ++g) {
      re1[blk][g] = (_Float16*)carve(512 * 256 * 2);
      re2[blk][g] = (_Float16*)carve(256 * 256 * 2);
      rn1[blk][g] = (_Float16*)carve(512 * 256 * 2);
      rn2[blk][g] = (_Float16*)carve(256 * 256 * 2);
    }
  }

  zero_kernel<<<1, 256, 0, stream>>>(velsum, 160);

  // Weight repack (first 512 rows of the 514-row matrices; tail handled in-kernel)
  for (int blk = 0; blk < 4; ++blk) {
    repack_kernel<<<16 * 32, 256, 0, stream>>>(c1w[blk], rc1[blk], 16);
    repack_kernel<<< 8 * 32, 256, 0, stream>>>(c2w[blk], rc2[blk], 8);
    for (int g = 0; g < 2; ++g) {
      repack_kernel<<<16 * 32, 256, 0, stream>>>(e1w[blk][g], re1[blk][g], 16);
      repack_kernel<<< 8 * 32, 256, 0, stream>>>(e2w[blk][g], re2[blk][g], 8);
      repack_kernel<<<16 * 32, 256, 0, stream>>>(n1w[blk][g], rn1[blk][g], 16);
      repack_kernel<<< 8 * 32, 256, 0, stream>>>(n2w[blk][g], rn2[blk][g], 8);
    }
  }

  embed_kernel<<<2048, 256, 0, stream>>>(t, xh, nm, emb_w, emb_b, hf, h16, x0, xA);

  float* xc = xA;
  float* xn = xB;
  for (int blk = 0; blk < 4; ++blk) {
    for (int g = 0; g < 2; ++g) {
      edge_gcl_kernel<<<2048, 256, 0, stream>>>(
          h16, xc, x0, em, re1[blk][g], e1w[blk][g] + 512 * 256, e1b[blk][g],
          re2[blk][g], e2b[blk][g], agg16);
      node_gcl_kernel<<<32, 256, 0, stream>>>(
          h16, agg16, nm, rn1[blk][g], n1b[blk][g], rn2[blk][g], n2b[blk][g],
          hf, h16);
    }
    coord_kernel<<<2048, 256, 0, stream>>>(
        h16, xc, x0, em, nm, rc1[blk], c1w[blk] + 512 * 256, c1b[blk],
        rc2[blk], c2b[blk], c3w[blk], xn);
    float* tmp = xc; xc = xn; xn = tmp;
  }

  final1_kernel<<<2048, 32, 0, stream>>>(hf, embo_w, embo_b, nm, xc, x0,
                                         hobuf, velraw, velsum, ncnt);
  final2_kernel<<<2048, 32, 0, stream>>>(hobuf, velraw, velsum, ncnt, nm,
                                         (float*)d_out);
}
